// ImageNavHFNetPolicyQ_53377853555342
// MI455X (gfx1250) — matrix operand order repaired and checked by reading, not compiled, not run on hardware
//
#include <hip/hip_runtime.h>
#include <hip/hip_bf16.h>

// ---------------------------------------------------------------------------
// Batched mutual-NN descriptor matching with ratio test (HFNet-style).
// B=32, N=1024, D=256, fp32 in. Fused: distances never materialized.
// Compute: 17.2 GFLOP einsum -> bf16 WMMA (v_wmma_f32_16x16x32_bf16),
// fp32 accumulate; row top-2 + packed-u64 column argmin fused into epilogue.
// B-chunks staged via async global->LDS copies (ASYNCcnt), double-buffered.
// ---------------------------------------------------------------------------

typedef __bf16 bf16_t;
typedef __attribute__((ext_vector_type(16))) __bf16 v16bf;
typedef __attribute__((ext_vector_type(8)))  float  v8f;
typedef __attribute__((ext_vector_type(4)))  int    v4i;

#define BQ 32
#define NQ 1024
#define DQ 256
#define ROWS_TOTAL (BQ * NQ)

#define AS_GLOBAL __attribute__((address_space(1)))
#define AS_LOCAL  __attribute__((address_space(3)))

// ---------------------------------------------------------------------------
// Kernel 1: fp32 -> bf16 conversion + row squared norms.
// One wave (32 lanes) per 256-elem row; lane handles 8 contiguous floats.
// ---------------------------------------------------------------------------
__global__ __launch_bounds__(256)
void prep_kernel(const float* __restrict__ d0, const float* __restrict__ d1,
                 bf16_t* __restrict__ h0, bf16_t* __restrict__ h1,
                 float* __restrict__ sq0, float* __restrict__ sq1) {
  const int wave = threadIdx.x >> 5;
  const int lane = threadIdx.x & 31;
  const int gRow = blockIdx.x * 8 + wave;          // 0 .. 2*B*N-1
  if (gRow >= 2 * ROWS_TOTAL) return;
  const bool is1 = gRow >= ROWS_TOTAL;
  const int  r   = is1 ? (gRow - ROWS_TOTAL) : gRow;

  const float* src = (is1 ? d1 : d0) + (size_t)r * DQ;
  bf16_t*      dst = (is1 ? h1 : h0) + (size_t)r * DQ;

  const float4 a = ((const float4*)src)[lane * 2 + 0];
  const float4 b = ((const float4*)src)[lane * 2 + 1];

  float s = a.x*a.x + a.y*a.y + a.z*a.z + a.w*a.w
          + b.x*b.x + b.y*b.y + b.z*b.z + b.w*b.w;

  union { bf16_t h[8]; uint4 u; } pk;
  pk.h[0] = (bf16_t)a.x; pk.h[1] = (bf16_t)a.y;
  pk.h[2] = (bf16_t)a.z; pk.h[3] = (bf16_t)a.w;
  pk.h[4] = (bf16_t)b.x; pk.h[5] = (bf16_t)b.y;
  pk.h[6] = (bf16_t)b.z; pk.h[7] = (bf16_t)b.w;
  ((uint4*)dst)[lane] = pk.u;

  #pragma unroll
  for (int off = 16; off >= 1; off >>= 1) s += __shfl_xor(s, off, 32);
  if (lane == 0) (is1 ? sq1 : sq0)[r] = s;
}

// ---------------------------------------------------------------------------
// Kernel 2: init packed column-min keys.
// ---------------------------------------------------------------------------
__global__ __launch_bounds__(256)
void init_colmin(unsigned long long* __restrict__ cm) {
  const int i = blockIdx.x * blockDim.x + threadIdx.x;
  if (i < ROWS_TOTAL) cm[i] = ~0ull;
}

// ---------------------------------------------------------------------------
// Async copy of one 32 KB B chunk (64 cols x 256 bf16) into LDS.
// 256 threads x 8 x b128 = 32 KB. Immediate offset applies to BOTH the
// global and LDS address per the ISA (global_load_async_to_lds_b128), and
// must be an integer-constant expression -> literal offsets.
// ---------------------------------------------------------------------------
__device__ __forceinline__
void async_copy_chunk(const bf16_t* __restrict__ gsrc, bf16_t* ldst, int tid) {
  AS_GLOBAL v4i* g = (AS_GLOBAL v4i*)((const char*)gsrc + tid * 16);
  AS_LOCAL  v4i* l = (AS_LOCAL  v4i*)((char*)ldst + tid * 16);
  __builtin_amdgcn_global_load_async_to_lds_b128(g, l,     0, 0);
  __builtin_amdgcn_global_load_async_to_lds_b128(g, l,  4096, 0);
  __builtin_amdgcn_global_load_async_to_lds_b128(g, l,  8192, 0);
  __builtin_amdgcn_global_load_async_to_lds_b128(g, l, 12288, 0);
  __builtin_amdgcn_global_load_async_to_lds_b128(g, l, 16384, 0);
  __builtin_amdgcn_global_load_async_to_lds_b128(g, l, 20480, 0);
  __builtin_amdgcn_global_load_async_to_lds_b128(g, l, 24576, 0);
  __builtin_amdgcn_global_load_async_to_lds_b128(g, l, 28672, 0);
}

__device__ __forceinline__ void wait_async0() {
  asm volatile("s_wait_asynccnt 0x0" ::: "memory");
}

// ---------------------------------------------------------------------------
// Kernel 3: fused WMMA distance + reductions.
// Grid: B * 8 workgroups (128 rows each), 256 threads = 8 waves.
// Wave w owns rows [rb*128 + w*16, +16), A stripe register-resident.
// desc1 streamed through double-buffered LDS via async copies.
// ---------------------------------------------------------------------------
__global__ __launch_bounds__(256, 1)
void match_kernel(const bf16_t* __restrict__ h0, const bf16_t* __restrict__ h1,
                  const float* __restrict__ sq0, const float* __restrict__ sq1,
                  unsigned long long* __restrict__ colMin,
                  int* __restrict__ rowIdx, float* __restrict__ rowBest,
                  float* __restrict__ rowSecond) {
  __shared__ bf16_t             ldsB[2][64 * DQ];  // 2 x 32 KB double buffer
  __shared__ unsigned long long ldsCol[NQ];        // 8 KB packed col mins

  const int b    = blockIdx.x >> 3;
  const int rb   = blockIdx.x & 7;
  const int wave = threadIdx.x >> 5;
  const int lane = threadIdx.x & 31;
  const int half = lane >> 4;       // 0: M/K-low half, 1: high half
  const int l16  = lane & 15;
  const int tid  = threadIdx.x;

  for (int i = tid; i < NQ; i += 256) ldsCol[i] = ~0ull;

  const int rowBase = rb * 128 + wave * 16;
  const bf16_t* A  = h0 + ((size_t)b * NQ + rowBase) * DQ;
  const bf16_t* Bg = h1 + (size_t)b * NQ * DQ;

  // Kick off chunk 0 while we set up A fragments.
  async_copy_chunk(Bg, &ldsB[0][0], tid);

  // A fragments for K=256 (8 ksteps of 32), 16-bit A 16x32 layout:
  // lane<16: M=lane, K = {k0..k0+7, k0+16..k0+23}; lane>=16: +8 on K.
  v16bf afrag[8];
  {
    const bf16_t* arow = A + l16 * DQ + (half ? 8 : 0);
    #pragma unroll
    for (int s = 0; s < 8; ++s) {
      union { v16bf v; uint4 u[2]; } f;
      f.u[0] = *(const uint4*)(arow + s * 32);
      f.u[1] = *(const uint4*)(arow + s * 32 + 16);
      afrag[s] = f.v;
    }
  }

  float sq0r[8];
  #pragma unroll
  for (int r = 0; r < 8; ++r)
    sq0r[r] = sq0[b * NQ + rowBase + r + 8 * half];

  float m1[8], m2[8]; int bi[8];
  #pragma unroll
  for (int r = 0; r < 8; ++r) {
    m1[r] = __builtin_inff(); m2[r] = __builtin_inff(); bi[r] = 0;
  }

  wait_async0();        // own share of chunk 0 landed
  __syncthreads();      // everyone's share landed (+ ldsCol init visible)

  for (int chunk = 0; chunk < 16; ++chunk) {
    const int c0  = chunk * 64;
    const int cur = chunk & 1;

    // Prefetch next chunk into the other buffer; overlaps with the WMMAs
    // below. Safe WAR: all waves passed the previous barrier, so nobody is
    // still reading buffer cur^1.
    if (chunk + 1 < 16)
      async_copy_chunk(Bg + (size_t)(c0 + 64) * DQ, &ldsB[cur ^ 1][0], tid);

    // Per-lane sq1 values for this chunk's 4 column tiles.
    float sq1v[4];
    #pragma unroll
    for (int ct = 0; ct < 4; ++ct)
      sq1v[ct] = sq1[b * NQ + c0 + ct * 16 + l16];

    for (int ct = 0; ct < 4; ++ct) {
      const int colT = ct * 16;
      const int col  = c0 + colT + l16;
      v8f acc = {};
      {
        // 16-bit B 32x16 layout: lane holds column N=l16, K = {k0..k0+7, k0+16..k0+23},
        // lanes>=16 take K+8. Row-major desc1 => contiguous LDS reads.
        const bf16_t* brow = &ldsB[cur][0] + (colT + l16) * DQ + (half ? 8 : 0);
        #pragma unroll
        for (int s = 0; s < 8; ++s) {
          union { v16bf v; uint4 u[2]; } f;
          f.u[0] = *(const uint4*)(brow + s * 32);
          f.u[1] = *(const uint4*)(brow + s * 32 + 16);
          acc = __builtin_amdgcn_wmma_f32_16x16x32_bf16(
              false, afrag[s], false, f.v, (short)0, acc, false, false);
        }
      }

      // Epilogue: d^2, per-row top-2, per-column packed argmin.
      const float sq1c = sq1v[ct];
      float cmin = __builtin_inff(); int crow = 0;
      #pragma unroll
      for (int r = 0; r < 8; ++r) {
        const float d2 = fmaxf(fmaf(-2.0f, acc[r], sq0r[r] + sq1c), 0.0f);
        const bool lt = d2 < m1[r];                 // strict: keeps first idx
        m2[r] = fminf(m2[r], fmaxf(d2, m1[r]));
        bi[r] = lt ? col : bi[r];
        m1[r] = fminf(m1[r], d2);
        const int  rw = rowBase + r + 8 * half;
        const bool cl = d2 < cmin;                  // strict: keeps low row
        crow = cl ? rw : crow;
        cmin = cl ? d2 : cmin;
      }
      const unsigned long long key =
          ((unsigned long long)__float_as_uint(cmin) << 32) | (unsigned)crow;
      atomicMin(&ldsCol[col], key);                 // ds_min_u64
    }

    wait_async0();      // own share of chunk+1 landed
    __syncthreads();    // all shares landed; all reads of buffer cur done
  }

  // Flush column mins to global (min dist, then min row, via u64 ordering).
  for (int i = tid; i < NQ; i += 256)
    atomicMin(&colMin[(size_t)b * NQ + i], ldsCol[i]);

  // Cross-lane top-2 merge within each 16-lane column group.
  #pragma unroll
  for (int off = 1; off < 16; off <<= 1) {
    #pragma unroll
    for (int r = 0; r < 8; ++r) {
      const float o1 = __shfl_xor(m1[r], off, 32);
      const float o2 = __shfl_xor(m2[r], off, 32);
      const int   oi = __shfl_xor(bi[r], off, 32);
      const float nm2 = fminf(fminf(m2[r], o2), fmaxf(m1[r], o1));
      const bool take = (o1 < m1[r]) || ((o1 == m1[r]) && (oi < bi[r]));
      m1[r] = take ? o1 : m1[r];
      bi[r] = take ? oi : bi[r];
      m2[r] = nm2;
    }
  }
  if (l16 == 0) {
    #pragma unroll
    for (int r = 0; r < 8; ++r) {
      const int row = rowBase + r + 8 * half;
      const size_t o = (size_t)b * NQ + row;
      rowBest[o]   = m1[r];   // squared distance; sqrt deferred
      rowSecond[o] = m2[r];
      rowIdx[o]    = bi[r];
    }
  }
}

// ---------------------------------------------------------------------------
// Kernel 4: cross-check + ratio test + output packing.
// d_out = [ (float)indices2 | (float)valid | neg_best ], each B*N.
// ---------------------------------------------------------------------------
__global__ __launch_bounds__(256)
void finalize_kernel(const unsigned long long* __restrict__ colMin,
                     const int* __restrict__ rowIdx,
                     const float* __restrict__ rowBest,
                     const float* __restrict__ rowSecond,
                     float* __restrict__ out) {
  const int i = blockIdx.x * blockDim.x + threadIdx.x;
  if (i >= ROWS_TOTAL) return;
  const int b = i >> 10;
  const int n = i & (NQ - 1);

  const int j = rowIdx[i];
  const unsigned long long pk = colMin[(size_t)b * NQ + j];
  const int mrow = (int)(unsigned)(pk & 0xffffffffull);

  const float best   = sqrtf(rowBest[i]);
  float       second = sqrtf(rowSecond[i]);
  const float EPSF = 1.1920928955078125e-07f;   // np.float32 eps
  second = (second == 0.0f) ? EPSF : second;

  const bool valid = (mrow == n) && (best < 0.8f * second);

  out[i]                  = (float)j;
  out[ROWS_TOTAL + i]     = valid ? 1.0f : 0.0f;
  out[2 * ROWS_TOTAL + i] = valid ? -best : 0.0f;
}

// ---------------------------------------------------------------------------
extern "C" void kernel_launch(void* const* d_in, const int* in_sizes, int n_in,
                              void* d_out, int out_size, void* d_ws, size_t ws_size,
                              hipStream_t stream) {
  const float* d0 = (const float*)d_in[0];
  const float* d1 = (const float*)d_in[1];
  float* out = (float*)d_out;

  char* ws = (char*)d_ws;
  size_t o = 0;
  bf16_t* h0 = (bf16_t*)(ws + o);               o += (size_t)ROWS_TOTAL * DQ * 2;
  bf16_t* h1 = (bf16_t*)(ws + o);               o += (size_t)ROWS_TOTAL * DQ * 2;
  float* sq0 = (float*)(ws + o);                o += (size_t)ROWS_TOTAL * 4;
  float* sq1 = (float*)(ws + o);                o += (size_t)ROWS_TOTAL * 4;
  unsigned long long* colMin = (unsigned long long*)(ws + o); o += (size_t)ROWS_TOTAL * 8;
  int*   rowIdx    = (int*)(ws + o);            o += (size_t)ROWS_TOTAL * 4;
  float* rowBest   = (float*)(ws + o);          o += (size_t)ROWS_TOTAL * 4;
  float* rowSecond = (float*)(ws + o);          o += (size_t)ROWS_TOTAL * 4;
  (void)o; (void)ws_size; (void)in_sizes; (void)n_in; (void)out_size;

  prep_kernel<<<(2 * ROWS_TOTAL) / 8, 256, 0, stream>>>(d0, d1, h0, h1, sq0, sq1);
  init_colmin<<<ROWS_TOTAL / 256, 256, 0, stream>>>(colMin);
  match_kernel<<<BQ * 8, 256, 0, stream>>>(h0, h1, sq0, sq1, colMin,
                                           rowIdx, rowBest, rowSecond);
  finalize_kernel<<<ROWS_TOTAL / 256, 256, 0, stream>>>(colMin, rowIdx, rowBest,
                                                        rowSecond, out);
}
